// MMOE_12816182411987
// MI455X (gfx1250) — matrix operand order repaired and checked by reading, not compiled, not run on hardware
//
#include <hip/hip_runtime.h>
#include <math.h>

typedef __attribute__((ext_vector_type(16))) _Float16 v16h;
typedef __attribute__((ext_vector_type(8)))  float    v8f;
typedef __attribute__((ext_vector_type(4)))  float    v4f;

#define BATCH   65536
#define DK      1024
#define NEXP    10
#define NUNIT   10
#define NCOL    128          // 120 real columns padded to 128
#define NKT     (DK / 32)    // 32 k-tiles of K=32
#define ROWS_PB 256          // rows per block (8 waves x 32 rows: 2 M-tiles/wave)
#define CSTR    132          // padded LDS C stride (floats) to avoid bank conflicts
#define LDS_BYTES (DK * NCOL * 2)  // 256 KB: packed f16 W; reused as f32 C tile

// ---------------------------------------------------------------------------
// Pack expert_w [E,D,U] and gate_w [T,D,E] (f32) into the per-lane WMMA
// B-fragment layout (f16), one v16h (32B) per (ktile, ntile, lane):
//   idx = ((kt*8 + nt)*32 + lane)*16 + i
//   k   = kt*32 + (i<8 ? (lane/16)*8 + i : 16 + (lane/16)*8 + (i-8))
//   n   = nt*16 + (lane%16)
// Column map: n<100 -> expert (e=n/10, u=n%10); 100<=n<120 -> gate (t,e); else 0.
// ---------------------------------------------------------------------------
__global__ void mmoe_pack_w(const float* __restrict__ expert_w,
                            const float* __restrict__ gate_w,
                            _Float16* __restrict__ wp) {
    int idx = blockIdx.x * blockDim.x + threadIdx.x;   // 0 .. 131071
    int i  = idx & 15;
    int l  = (idx >> 4) & 31;
    int nt = (idx >> 9) & 7;
    int kt = idx >> 12;
    int k  = kt * 32 + ((i < 8) ? (l >> 4) * 8 + i : 16 + (l >> 4) * 8 + (i - 8));
    int n  = nt * 16 + (l & 15);
    float v = 0.0f;
    if (n < 100) {
        int e = n / 10, u = n % 10;
        v = expert_w[(size_t)e * DK * NUNIT + (size_t)k * NUNIT + u];
    } else if (n < 120) {
        int t = (n - 100) / 10, e = (n - 100) % 10;
        v = gate_w[(size_t)t * DK * NEXP + (size_t)k * NEXP + e];
    }
    wp[idx] = (_Float16)v;
}

// Load one 16x32 f16 A fragment (ISA 16-bit A layout) from an f32 row.
//   lane halfs 0..7  -> K = kb + j        (kb = kt*32 + half*8)
//   lane halfs 8..15 -> K = kb + 16 + j
__device__ __forceinline__ v16h load_a_frag(const float* __restrict__ xrow, int kb) {
    v4f a0 = *(const v4f*)(xrow + kb);
    v4f a1 = *(const v4f*)(xrow + kb + 4);
    v4f a2 = *(const v4f*)(xrow + kb + 16);
    v4f a3 = *(const v4f*)(xrow + kb + 20);
    v16h A;
#pragma unroll
    for (int j = 0; j < 4; ++j) {
        A[j]      = (_Float16)a0[j];
        A[4 + j]  = (_Float16)a1[j];
        A[8 + j]  = (_Float16)a2[j];
        A[12 + j] = (_Float16)a3[j];
    }
    return A;
}

// ---------------------------------------------------------------------------
// Fused MMoE: C = X * W (f16 WMMA, f32 accum), then relu/softmax/mixture/
// sigmoid epilogue. One block = 256 rows; whole W lives in LDS; each wave
// runs two M-tiles so every B fragment from LDS feeds two WMMAs.
// ---------------------------------------------------------------------------
__global__ void __launch_bounds__(256)
mmoe_main(const float* __restrict__ x,
          const _Float16* __restrict__ wp,
          const float* __restrict__ expert_b,
          const float* __restrict__ gate_b,
          const float* __restrict__ ctr_w, const float* __restrict__ ctr_b,
          const float* __restrict__ cvr_w, const float* __restrict__ cvr_b,
          float* __restrict__ out) {
    extern __shared__ char smem[];
    const int tid  = threadIdx.x;
    const int wave = tid >> 5;
    const int lane = tid & 31;

    // ---- stage packed W (256 KB) into LDS ----
    {
        const uint4* src = (const uint4*)wp;
        uint4*       dst = (uint4*)smem;
        for (int j = tid; j < LDS_BYTES / 16; j += 256) dst[j] = src[j];
    }
    __syncthreads();

    // ---- GEMM phase: 2 M-tiles (32 rows) per wave ----
    v8f acc0[8] = {};
    v8f acc1[8] = {};
    const int m    = lane & 15;   // row within 16-row tile
    const int half = lane >> 4;   // which K-half this lane supplies
    const size_t row0 = (size_t)blockIdx.x * ROWS_PB + wave * 32 + m;
    const float* xrow0 = x + row0 * DK;
    const float* xrow1 = xrow0 + (size_t)16 * DK;
    const v16h* ldsB = (const v16h*)smem;

    for (int kt = 0; kt < NKT; ++kt) {
        const int kb = kt * 32 + half * 8;
        // keep the x stream ahead of the WMMA pipe (global_prefetch_b8)
        __builtin_prefetch(xrow0 + kb + 8 * 32, 0, 0);
        __builtin_prefetch(xrow1 + kb + 8 * 32, 0, 0);
        v16h A0 = load_a_frag(xrow0, kb);
        v16h A1 = load_a_frag(xrow1, kb);
#pragma unroll
        for (int nt = 0; nt < 8; ++nt) {
            v16h Bf = ldsB[(kt * 8 + nt) * 32 + lane];  // one LDS read, two WMMAs
            acc0[nt] = __builtin_amdgcn_wmma_f32_16x16x32_f16(
                false, A0, false, Bf, (short)0, acc0[nt], false, false);
            acc1[nt] = __builtin_amdgcn_wmma_f32_16x16x32_f16(
                false, A1, false, Bf, (short)0, acc1[nt], false, false);
        }
    }

    // ---- spill C tiles to LDS (reuse W region) ----
    __syncthreads();
    float* ldsC = (float*)smem;
#pragma unroll
    for (int nt = 0; nt < 8; ++nt)
#pragma unroll
        for (int r = 0; r < 8; ++r) {
            ldsC[(wave * 32 + r + 8 * half) * CSTR + nt * 16 + m]      = acc0[nt][r];
            ldsC[(wave * 32 + 16 + r + 8 * half) * CSTR + nt * 16 + m] = acc1[nt][r];
        }
    __syncthreads();

    // ---- epilogue: one thread per row (256 threads <-> 256 rows) ----
    {
        const float* c = ldsC + tid * CSTR;
        const size_t grow = (size_t)blockIdx.x * ROWS_PB + tid;
#pragma unroll
        for (int t = 0; t < 2; ++t) {
            // softmax over experts for task t (columns 100 + t*10 + e)
            float g[NEXP];
            float mx = -1e30f;
#pragma unroll
            for (int e = 0; e < NEXP; ++e) {
                g[e] = c[100 + t * 10 + e] + gate_b[t * NEXP + e];
                mx = fmaxf(mx, g[e]);
            }
            float s = 0.0f;
#pragma unroll
            for (int e = 0; e < NEXP; ++e) { g[e] = __expf(g[e] - mx); s += g[e]; }
            const float inv = 1.0f / s;

            float logit = (t == 0) ? ctr_b[0] : cvr_b[0];
#pragma unroll
            for (int u = 0; u < NUNIT; ++u) {
                float ti = 0.0f;
#pragma unroll
                for (int e = 0; e < NEXP; ++e)
                    ti += fmaxf(c[e * 10 + u] + expert_b[e * NUNIT + u], 0.0f) * g[e];
                ti *= inv;
                logit += ti * ((t == 0) ? ctr_w[u] : cvr_w[u]);
            }
            out[(size_t)t * BATCH + grow] = 1.0f / (1.0f + __expf(-logit));
        }
    }
}

extern "C" void kernel_launch(void* const* d_in, const int* in_sizes, int n_in,
                              void* d_out, int out_size, void* d_ws, size_t ws_size,
                              hipStream_t stream) {
    (void)in_sizes; (void)n_in; (void)out_size; (void)ws_size;
    const float* x        = (const float*)d_in[0];
    // d_in[1] show_index, d_in[2] st: unused by the reference outputs
    const float* expert_w = (const float*)d_in[3];
    const float* expert_b = (const float*)d_in[4];
    const float* gate_w   = (const float*)d_in[5];
    const float* gate_b   = (const float*)d_in[6];
    const float* ctr_w    = (const float*)d_in[7];
    const float* ctr_b    = (const float*)d_in[8];
    const float* cvr_w    = (const float*)d_in[9];
    const float* cvr_b    = (const float*)d_in[10];
    float* out = (float*)d_out;
    _Float16* wp = (_Float16*)d_ws;   // 256 KB packed W

    // 1) pack W into WMMA B-fragment layout (131072 halves)
    mmoe_pack_w<<<512, 256, 0, stream>>>(expert_w, gate_w, wp);

    // 2) fused GEMM + epilogue; 256 KB dynamic LDS (whole W resident per WGP)
    hipFuncSetAttribute(reinterpret_cast<const void*>(mmoe_main),
                        hipFuncAttributeMaxDynamicSharedMemorySize, LDS_BYTES);
    mmoe_main<<<BATCH / ROWS_PB, 256, LDS_BYTES, stream>>>(
        x, wp, expert_b, gate_b, ctr_w, ctr_b, cvr_w, cvr_b, out);
}
